// SelfAttention_31963146617231
// MI455X (gfx1250) — compile-verified, matrix operand order checked by reading, not run on hardware
//
#include <hip/hip_runtime.h>

#define BATCH 4
#define CDIM  256
#define HWDIM 4096
#define DKDIM 32

typedef __attribute__((ext_vector_type(16))) _Float16 v16h;
typedef __attribute__((ext_vector_type(8)))  _Float16 v8h;
typedef __attribute__((ext_vector_type(8)))  float    v8f;

// ---------------------------------------------------------------------------
// Load a 16x32 (rows x K) f16 operand fragment in the CDNA5 WMMA A/B layout.
// Per CDNA5 ISA (16-bit A-matrix 16x32):
//   lanes 0-15 : row = lane,    K = 0..7 (v0-3) and 16..23 (v4-7)
//   lanes 16-31: row = lane-16, K = 8..15       and 24..31
// Row data must be contiguous with `stride` halfs between rows.
// B fragments use the same per-lane mapping with "row" = column index.
// ---------------------------------------------------------------------------
__device__ __forceinline__ v16h load_tile32(const _Float16* base, int stride,
                                            int l, int half) {
    const _Float16* p = base + (size_t)l * stride + half * 8;
    v8h lo = *(const v8h*)(p);        // K =  0..7  or  8..15
    v8h hi = *(const v8h*)(p + 16);   // K = 16..23 or 24..31
    v16h r;
#pragma unroll
    for (int i = 0; i < 8; ++i) { r[i] = lo[i]; r[i + 8] = hi[i]; }
    return r;
}

// lane <-> lane^16 exchange as a VALU permlane op (identity selects), avoiding
// the ds_bpermute + s_wait_dscnt round-trip that __shfl_xor lowers to.
__device__ __forceinline__ float xhalf(float v) {
    int i = __float_as_int(v);
    int o = __builtin_amdgcn_permlanex16(i, i, 0x76543210, 0xfedcba98, true, false);
    return __int_as_float(o);
}

// ---------------------------------------------------------------------------
// 1x1-conv projection: out[oc, n] = bias[oc] + sum_c w[oc,c] * x[b,c,n]
// layoutT == 0 : out stored [b][n][outc]  -> Q, K   (row n contiguous over d)
// layoutT == 1 : out stored [b][outc][n]  -> V      (row e contiguous over n)
// ---------------------------------------------------------------------------
__global__ __launch_bounds__(256) void proj_kernel(
    const float* __restrict__ x, const float* __restrict__ w,
    const float* __restrict__ bias, _Float16* __restrict__ out,
    int outc, int layoutT)
{
    int blk   = blockIdx.x;
    int chunk = blk & 15;                 // 16 chunks of 256 over HW=4096
    int oc    = (blk >> 4) % outc;
    int b     = blk / (outc * 16);
    int n     = chunk * 256 + threadIdx.x;

    const float* xp = x + (size_t)b * CDIM * HWDIM + n;
    const float* wp = w + (size_t)oc * CDIM;
    float acc = bias[oc];
#pragma unroll 8
    for (int c = 0; c < CDIM; ++c)
        acc = fmaf(wp[c], xp[(size_t)c * HWDIM], acc);

    if (layoutT)
        out[((size_t)b * outc + oc) * HWDIM + n] = (_Float16)acc;
    else
        out[((size_t)b * HWDIM + n) * outc + oc] = (_Float16)acc;
}

// ---------------------------------------------------------------------------
// Fused flash-attention, transposed formulation:
//   S^T = K_tile (A) x Q_cols (B)      -> lane l owns query l's scores in regs
//   O^T[t] = V_tile (A) x P^T_cols (B) -> lane l owns query l's output column
//
// Key property: the C/D layout of S^T (lane l, half h holds keys
// {r + h*8} and {16 + r + h*8} for r=0..7) is IDENTICAL to the B-fragment
// K-mapping required for P^T (half 0: K 0-7/16-23, half 1: K 8-15/24-31).
// So P^T is repacked f32->f16 entirely in registers — zero LDS traffic, no
// cross-lane movement between the two WMMA stages.
// ---------------------------------------------------------------------------
__global__ __launch_bounds__(128) void attn_kernel(
    const float*    __restrict__ x,
    const _Float16* __restrict__ Qh,   // [b][n][dk]
    const _Float16* __restrict__ Kh,   // [b][n][dk]
    const _Float16* __restrict__ Vh,   // [b][e][n]
    const float*    __restrict__ gamma,
    float*          __restrict__ out)
{
    const int lane = threadIdx.x & 31;
    const int wid  = threadIdx.x >> 5;
    const int l    = lane & 15;
    const int half = lane >> 4;
    const int tile = blockIdx.x * 4 + wid;      // 1024 query tiles total
    const int b    = tile >> 8;                 // 256 tiles per batch
    const int m0   = (tile & 255) << 4;

    // Q as B-matrix (32 dk x 16 queries): lane holds column q = m0 + l.
    // Loaded ONCE — reused by every score WMMA.
    v16h bQ = load_tile32(Qh + ((size_t)b * HWDIM + m0) * DKDIM, DKDIM, l, half);

    // O^T accumulators: tile t rows e = t*16 + r + half*8, col m = m0 + l
    v8f acc[16];
#pragma unroll
    for (int t = 0; t < 16; ++t) acc[t] = (v8f){};

    float rowm = -INFINITY;     // running max for query m0 + l
    float rsum = 0.0f;          // running denominator

    for (int n0 = 0; n0 < HWDIM; n0 += 32) {
        // K rows as A-matrix tiles (16 keys x 32 dk each)
        v16h aK0 = load_tile32(Kh + ((size_t)b * HWDIM + n0)      * DKDIM, DKDIM, l, half);
        v16h aK1 = load_tile32(Kh + ((size_t)b * HWDIM + n0 + 16) * DKDIM, DKDIM, l, half);

        v8f z = {};
        // s0[r]: key n0 +      r + half*8, query m0 + l
        // s1[r]: key n0 + 16 + r + half*8, query m0 + l
        v8f s0 = __builtin_amdgcn_wmma_f32_16x16x32_f16(false, aK0, false, bQ,
                                                        (short)0, z, false, false);
        v8f s1 = __builtin_amdgcn_wmma_f32_16x16x32_f16(false, aK1, false, bQ,
                                                        (short)0, z, false, false);

        // block max over this lane's 16 keys (register-axis reduction)
        float bm = fmaxf(s0[0], s1[0]);
#pragma unroll
        for (int r = 1; r < 8; ++r) bm = fmaxf(bm, fmaxf(s0[r], s1[r]));
        bm = fmaxf(bm, xhalf(bm));                 // combine the two halves

        float nm    = fmaxf(rowm, bm);
        float alpha = __expf(rowm - nm);
        rowm = nm;

        float bs = 0.0f;
#pragma unroll
        for (int r = 0; r < 8; ++r) {
            s0[r] = __expf(s0[r] - nm);
            s1[r] = __expf(s1[r] - nm);
            bs += s0[r] + s1[r];
        }
        bs += xhalf(bs);                           // combine the two halves
        rsum = rsum * alpha + bs;

        // Repack P^T as the B fragment directly in registers:
        //   elements 0..7  = s0 (K =  0..7  /  8..15)
        //   elements 8..15 = s1 (K = 16..23 / 24..31)
        v16h bP;
#pragma unroll
        for (int r = 0; r < 8; ++r) {
            bP[r]     = (_Float16)s0[r];
            bP[r + 8] = (_Float16)s1[r];
        }

        // rescale running accumulators (alpha is a per-lane scalar)
#pragma unroll
        for (int t = 0; t < 16; ++t)
#pragma unroll
            for (int r = 0; r < 8; ++r)
                acc[t][r] *= alpha;

        // accumulate V (A) x P^T (B) over all 16 channel tiles
        const _Float16* vbase = Vh + (size_t)b * CDIM * HWDIM + n0;
#pragma unroll
        for (int t = 0; t < 16; ++t) {
            v16h aV = load_tile32(vbase + (size_t)(t * 16) * HWDIM, HWDIM, l, half);
            acc[t] = __builtin_amdgcn_wmma_f32_16x16x32_f16(false, aV, false, bP,
                                                            (short)0, acc[t],
                                                            false, false);
        }
    }

    // epilogue: out = gamma * (O / rowsum) + x ; lane owns column m = m0 + l
    float g   = gamma[0];
    float inv = 1.0f / rsum;
    int   m   = m0 + l;
#pragma unroll
    for (int t = 0; t < 16; ++t) {
#pragma unroll
        for (int r = 0; r < 8; ++r) {
            int e = t * 16 + r + half * 8;
            size_t idx = ((size_t)b * CDIM + e) * HWDIM + m;
            out[idx] = g * acc[t][r] * inv + x[idx];
        }
    }
}

// ---------------------------------------------------------------------------
extern "C" void kernel_launch(void* const* d_in, const int* in_sizes, int n_in,
                              void* d_out, int out_size, void* d_ws, size_t ws_size,
                              hipStream_t stream) {
    const float* x     = (const float*)d_in[0];
    const float* wq    = (const float*)d_in[1];
    const float* bq    = (const float*)d_in[2];
    const float* wk    = (const float*)d_in[3];
    const float* bk    = (const float*)d_in[4];
    const float* wv    = (const float*)d_in[5];
    const float* bv    = (const float*)d_in[6];
    const float* gamma = (const float*)d_in[7];
    float* out = (float*)d_out;

    // workspace: Qh (1 MB) | Kh (1 MB) | Vh (8 MB)
    _Float16* Qh = (_Float16*)d_ws;
    _Float16* Kh = Qh + (size_t)BATCH * HWDIM * DKDIM;
    _Float16* Vh = Kh + (size_t)BATCH * HWDIM * DKDIM;

    proj_kernel<<<BATCH * DKDIM * 16, 256, 0, stream>>>(x, wq, bq, Qh, DKDIM, 0);
    proj_kernel<<<BATCH * DKDIM * 16, 256, 0, stream>>>(x, wk, bk, Kh, DKDIM, 0);
    proj_kernel<<<BATCH * CDIM  * 16, 256, 0, stream>>>(x, wv, bv, Vh, CDIM, 1);

    // 1024 query tiles, 4 waves (tiles) per 128-thread block
    attn_kernel<<<(BATCH * HWDIM / 16) / 4, 128, 0, stream>>>(x, Qh, Kh, Vh,
                                                              gamma, out);
}
